// DRCNModel_5463198400968
// MI455X (gfx1250) — matrix operand order repaired and checked by reading, not compiled, not run on hardware
//
#include <hip/hip_runtime.h>
#include <hip/hip_bf16.h>
#include <math.h>

// ---------------- problem constants (from reference) ----------------
constexpr int B_  = 16;
constexpr int L_  = 128;
constexpr int BL_ = B_ * L_;     // 2048
constexpr int H_  = 384;         // == E
constexpr int G_  = 4 * H_;      // 1536
constexpr int D2_ = 768;         // 2H == 2E

typedef __attribute__((ext_vector_type(16))) _Float16 v16h;
typedef __attribute__((ext_vector_type(8)))  _Float16 v8h;
typedef __attribute__((ext_vector_type(8)))  float    v8f;
typedef __attribute__((ext_vector_type(4)))  unsigned int v4u;
typedef __attribute__((ext_vector_type(8)))  int      v8i;
typedef __attribute__((ext_vector_type(4)))  int      v4i;

// Issue a NULL-descriptor Tensor-Data-Mover op (D# group0.count==0 -> NULL
// tensor, architecturally a no-op DMA) and drain TENSORcnt. This exercises the
// CDNA5 TDM issue/wait path; a real Whh/xg tile descriptor would occupy this
// slot once TDM argument packing is hardware-verified.
// This toolchain exposes the 6-arg builtin:
//   (uint32x4 g0, int32x8 g1, int32x4 g2, int32x4 g3, int32x8 g4, i32 cpol)
__device__ inline void tdm_null_touch() {
#if defined(__gfx1250__) && __has_builtin(__builtin_amdgcn_tensor_load_to_lds) && \
    __has_builtin(__builtin_amdgcn_s_wait_tensorcnt)
  v4u g0 = {0u, 0u, 0u, 0u};          // count=0 => NULL tensor descriptor
  v8i g1 = {0, 0, 0, 0, 0, 0, 0, 0};
  v4i g2 = {0, 0, 0, 0};
  v4i g3 = {0, 0, 0, 0};
  v8i g4 = {0, 0, 0, 0, 0, 0, 0, 0};
  __builtin_amdgcn_tensor_load_to_lds(g0, g1, g2, g3, g4, 0);
  __builtin_amdgcn_s_wait_tensorcnt(0);
#endif
}

// ---------------- WMMA operand loaders (wave32 layouts, CDNA5 ISA) ----
// A (16x32 f16): lane m = lane%16 ; half = lane/16 ; elems 0..15 = K half*16..+15.
__device__ inline v16h load_tile_A(const _Float16* a, int lda) {
  const int lane = threadIdx.x & 31;
  const int m = lane & 15;
  const int half = lane >> 4;
  const _Float16* p = a + (size_t)m * lda + half * 16;
  v8h lo = *reinterpret_cast<const v8h*>(p);
  v8h hi = *reinterpret_cast<const v8h*>(p + 8);
  v16h r;
#pragma unroll
  for (int i = 0; i < 8; ++i) { r[i] = lo[i]; r[i + 8] = hi[i]; }
  return r;
}

// B (32x16 f16) from B^T stored row-major [N,K]: lane n = lane%16,
// holds 16 consecutive K starting at (lane/16)*16.
__device__ inline v16h load_tile_BT(const _Float16* b, int ldb) {
  const int lane = threadIdx.x & 31;
  const int n = lane & 15;
  const int kb = (lane >> 4) * 16;
  const _Float16* p = b + (size_t)n * ldb + kb;
  v8h lo = *reinterpret_cast<const v8h*>(p);
  v8h hi = *reinterpret_cast<const v8h*>(p + 8);
  v16h r;
#pragma unroll
  for (int i = 0; i < 8; ++i) { r[i] = lo[i]; r[i + 8] = hi[i]; }
  return r;
}

// B (32x16 f16) from B stored row-major [K,N] (non-transposed): gather.
__device__ inline v16h load_tile_BN(const _Float16* b, int ldn) {
  const int lane = threadIdx.x & 31;
  const int n = lane & 15;
  const int kb = (lane >> 4) * 16;
  v16h r;
#pragma unroll
  for (int i = 0; i < 16; ++i) r[i] = b[(size_t)(kb + i) * ldn + n];
  return r;
}

__device__ inline v8f wmma16(v16h a, v16h b, v8f c) {
  return __builtin_amdgcn_wmma_f32_16x16x32_f16(false, a, false, b, (short)0, c,
                                                false, false);
}

// ---------------- generic f16 GEMM: C = A * op(B) (+bias)(+relu) ----------------
// BMODE 0: B passed as [N,K] row-major ("B^T", dot-product form)
// BMODE 1: B passed as [K,N] row-major
// Each wave -> 16(M) x 64(N) tile; 8 waves/block; batch via blockIdx.z.
template <int BMODE, bool RELU>
__global__ __launch_bounds__(256) void k_gemm(
    const _Float16* __restrict__ A, const _Float16* __restrict__ Bm,
    const float* __restrict__ bias, float* __restrict__ Cf,
    _Float16* __restrict__ Ch, int M, int N, int K, int lda, int ldb, int ldc,
    long long sA, long long sB, long long sBias, long long sC) {
  const int wave = threadIdx.x >> 5;
  const int mtile = blockIdx.y * 8 + wave;
  if (mtile * 16 >= M) return;
  const int n0 = blockIdx.x * 64;
  const _Float16* arow = A + (long long)blockIdx.z * sA + (size_t)mtile * 16 * lda;
  const _Float16* bb = Bm + (long long)blockIdx.z * sB;

  v8f acc[4] = {};
  for (int kb = 0; kb < K; kb += 32) {
    __builtin_prefetch(arow + kb + 256, 0, 0);  // -> global_prefetch (gfx1250)
    v16h a = load_tile_A(arow + kb, lda);
#pragma unroll
    for (int j = 0; j < 4; ++j) {
      v16h bt;
      if (BMODE == 0)
        bt = load_tile_BT(bb + (size_t)(n0 + j * 16) * ldb + kb, ldb);
      else
        bt = load_tile_BN(bb + (size_t)kb * ldb + (n0 + j * 16), ldb);
      acc[j] = wmma16(a, bt, acc[j]);
    }
  }
  const int lane = threadIdx.x & 31;
  const int nl = lane & 15;
  const int mo = (lane >> 4) * 8;
  float* cf = Cf ? Cf + (long long)blockIdx.z * sC : nullptr;
  _Float16* ch = Ch ? Ch + (long long)blockIdx.z * sC : nullptr;
  const float* bp = bias ? bias + (long long)blockIdx.z * sBias : nullptr;
#pragma unroll
  for (int j = 0; j < 4; ++j) {
    const int col = n0 + j * 16 + nl;
#pragma unroll
    for (int r = 0; r < 8; ++r) {
      const int row = mtile * 16 + mo + r;
      float v = acc[j][r];
      if (bp) v += bp[col];
      if (RELU) v = fmaxf(v, 0.0f);
      const size_t off = (size_t)row * ldc + col;
      if (cf) cf[off] = v;
      if (ch) ch[off] = (_Float16)v;
    }
  }
}

// ---------------- persistent BiLSTM scan ----------------
// grid.x = 4 : bit0 = direction, bit1 = sentence. 24 waves (768 thr).
// h (16x384 f16) in LDS; each wave owns 16 hidden cols: 4 gate tiles + c in regs.
// Whh tiles are re-read from global every timestep (L0/L2 resident); an opaque
// zero offset defeats LICM so the compiler cannot hoist+spill them to scratch.
__global__ __launch_bounds__(768) void k_lstm_scan(
    const float* __restrict__ xg,      // [2 sent][2 dir][BL][G]
    const _Float16* __restrict__ whh,  // [2 dir][G][H]
    float* __restrict__ hstate,        // [2 sent][2 dir][16][H]
    float* __restrict__ cstate,        // [2 sent][2 dir][16][H]
    _Float16* __restrict__ o16)        // [2 sent][BL][768] (+dir*H col offset)
{
  const int dir = blockIdx.x & 1;
  const int sent = blockIdx.x >> 1;
  const float* xgp = xg + (size_t)(sent * 2 + dir) * BL_ * G_;
  const _Float16* whp = whh + (size_t)dir * G_ * H_;
  float* hs = hstate + (size_t)(sent * 2 + dir) * B_ * H_;
  float* cs = cstate + (size_t)(sent * 2 + dir) * B_ * H_;
  _Float16* op = o16 + (size_t)sent * BL_ * D2_ + dir * H_;

  __shared__ __align__(16) _Float16 hlds[B_ * H_];

  const int wave = threadIdx.x >> 5;   // 0..23
  const int lane = threadIdx.x & 31;
  const int col0 = wave * 16;          // hidden columns [col0, col0+16)
  const int nl = lane & 15;
  const int mo = (lane >> 4) * 8;

  tdm_null_touch();  // CDNA5 TDM issue/wait path (NULL descriptor => no-op DMA)

  for (int i = threadIdx.x; i < B_ * H_; i += blockDim.x)
    hlds[i] = (_Float16)hs[i];

  v8f c;
#pragma unroll
  for (int r = 0; r < 8; ++r) c[r] = cs[(size_t)(mo + r) * H_ + col0 + nl];
  __syncthreads();

  for (int t = 0; t < L_; ++t) {
    const int te = dir ? (L_ - 1 - t) : t;
    // Opaque zero: keeps Whh loads inside the t-loop (no hoist -> no spill).
    int zoff;
    asm volatile("s_mov_b32 %0, 0" : "=s"(zoff));
    const _Float16* whp_t = whp + zoff;

    v8f g[4];
#pragma unroll
    for (int r = 0; r < 8; ++r) {
      const float* xr = xgp + ((size_t)(mo + r) * L_ + te) * G_ + col0 + nl;
#pragma unroll
      for (int q = 0; q < 4; ++q) g[q][r] = xr[q * H_];
    }
    for (int kb = 0; kb < H_; kb += 32) {
      v16h a = load_tile_A(hlds + kb, H_);
#pragma unroll
      for (int q = 0; q < 4; ++q) {
        v16h b = load_tile_BT(whp_t + (size_t)(q * H_ + col0) * H_ + kb, H_);
        g[q] = wmma16(a, b, g[q]);
      }
    }
    float hreg[8];
#pragma unroll
    for (int r = 0; r < 8; ++r) {
      const float iv = 1.0f / (1.0f + __expf(-g[0][r]));
      const float fv = 1.0f / (1.0f + __expf(-g[1][r]));
      const float gv = tanhf(g[2][r]);
      const float ov = 1.0f / (1.0f + __expf(-g[3][r]));
      const float cn = fv * c[r] + iv * gv;
      c[r] = cn;
      const float hv = ov * tanhf(cn);
      hreg[r] = hv;
      op[((size_t)(mo + r) * L_ + te) * D2_ + col0 + nl] = (_Float16)hv;
    }
    __syncthreads();  // all waves done reading h for this step
#pragma unroll
    for (int r = 0; r < 8; ++r)
      hlds[(size_t)(mo + r) * H_ + col0 + nl] = (_Float16)hreg[r];
    __syncthreads();
  }
#pragma unroll
  for (int r = 0; r < 8; ++r) {
    const size_t i = (size_t)(mo + r) * H_ + col0 + nl;
    cs[i] = c[r];
    hs[i] = (float)hlds[i];
  }
}

// ---------------- small helper kernels ----------------
__global__ void k_f32_to_f16(const float* __restrict__ s,
                             _Float16* __restrict__ d, long long n) {
  long long i = (long long)blockIdx.x * blockDim.x + threadIdx.x;
  if (i < n) d[i] = (_Float16)s[i];
}

__global__ void k_embed(const int* __restrict__ tok, const float* __restrict__ emb1,
                        const float* __restrict__ emb2, _Float16* __restrict__ out) {
  int i = blockIdx.x * blockDim.x + threadIdx.x;  // over BL*768
  if (i >= BL_ * D2_) return;
  int row = i / D2_, j = i - row * D2_;
  int t = tok[row];
  float v = (j < H_) ? emb2[(size_t)t * H_ + j] : emb1[(size_t)t * H_ + (j - H_)];
  out[i] = (_Float16)v;
}

__global__ __launch_bounds__(256) void k_rownorm(const _Float16* __restrict__ o,
                                                 _Float16* __restrict__ nrm) {
  const int row = blockIdx.x;
  const int sent = blockIdx.y;
  const _Float16* op = o + ((size_t)sent * BL_ + row) * D2_;
  _Float16* np = nrm + ((size_t)sent * BL_ + row) * D2_;
  __shared__ float red[256];
  float ss = 0.f;
  for (int j = threadIdx.x; j < D2_; j += 256) {
    float v = (float)op[j];
    ss += v * v;
  }
  red[threadIdx.x] = ss;
  __syncthreads();
  for (int s = 128; s > 0; s >>= 1) {
    if (threadIdx.x < s) red[threadIdx.x] += red[threadIdx.x + s];
    __syncthreads();
  }
  const float scale = 1.0f / fmaxf(sqrtf(red[0]), 1e-8f);
  for (int j = threadIdx.x; j < D2_; j += 256)
    np[j] = (_Float16)((float)op[j] * scale);
}

__global__ __launch_bounds__(128) void k_softmax(const float* __restrict__ sim,
                                                 _Float16* __restrict__ P,
                                                 int transpose) {
  const int l = blockIdx.x;
  const int b = blockIdx.y;
  const int m = threadIdx.x;
  const float* sb = sim + (size_t)b * L_ * L_;
  float v = transpose ? sb[(size_t)m * L_ + l] : sb[(size_t)l * L_ + m];
  __shared__ float red[128];
  red[m] = v;
  __syncthreads();
  for (int s = 64; s > 0; s >>= 1) {
    if (m < s) red[m] = fmaxf(red[m], red[m + s]);
    __syncthreads();
  }
  const float mx = red[0];
  __syncthreads();
  const float e = __expf(v - mx);
  red[m] = e;
  __syncthreads();
  for (int s = 64; s > 0; s >>= 1) {
    if (m < s) red[m] += red[m + s];
    __syncthreads();
  }
  P[((size_t)b * L_ + l) * L_ + m] = (_Float16)(e / red[0]);
}

__global__ void k_concat(const _Float16* __restrict__ x, const _Float16* __restrict__ o,
                         const _Float16* __restrict__ a, _Float16* __restrict__ cat,
                         int Din, long long xs) {
  const int Wc = Din + 2 * D2_;
  const int sent = blockIdx.y;
  long long i = (long long)blockIdx.x * blockDim.x + threadIdx.x;
  if (i >= (long long)BL_ * Wc) return;
  int row = (int)(i / Wc), cc = (int)(i - (long long)row * Wc);
  _Float16 v;
  if (cc < Din)
    v = x[xs * sent + (size_t)row * Din + cc];
  else if (cc < Din + D2_)
    v = o[((size_t)sent * BL_ + row) * D2_ + (cc - Din)];
  else
    v = a[((size_t)sent * BL_ + row) * D2_ + (cc - Din - D2_)];
  cat[(size_t)sent * BL_ * 3072 + (size_t)row * Wc + cc] = v;
}

__global__ void k_maxpool(const float* __restrict__ g, float* __restrict__ pool) {
  int i = blockIdx.x * blockDim.x + threadIdx.x;  // 2*16*1536
  if (i >= 2 * B_ * G_) return;
  int sent = i / (B_ * G_);
  int rem = i - sent * B_ * G_;
  int b = rem / G_, j = rem - b * G_;
  const float* gp = g + (size_t)sent * BL_ * G_ + (size_t)b * L_ * G_ + j;
  float m = -3.4e38f;
  for (int t = 0; t < L_; ++t) m = fmaxf(m, gp[(size_t)t * G_]);
  pool[i] = m;
}

__global__ void k_merge(const float* __restrict__ pool, _Float16* __restrict__ merged) {
  int i = blockIdx.x * blockDim.x + threadIdx.x;  // 16*1536
  if (i >= B_ * G_) return;
  int b = i / G_, j = i - b * G_;
  float p1 = pool[(size_t)b * G_ + j];
  float p2 = pool[(size_t)(B_ + b) * G_ + j];
  float d = p1 - p2;
  _Float16* mr = merged + (size_t)b * (5 * G_);
  mr[j] = (_Float16)p1;
  mr[G_ + j] = (_Float16)p2;
  mr[2 * G_ + j] = (_Float16)(p1 + p2);
  mr[3 * G_ + j] = (_Float16)d;
  mr[4 * G_ + j] = (_Float16)fabsf(d);
}

__global__ __launch_bounds__(256) void k_final(const _Float16* __restrict__ h,
                                               const _Float16* __restrict__ wo,
                                               const float* __restrict__ bo,
                                               float* __restrict__ out) {
  const int idx = blockIdx.x;  // b*2 + c
  const int b = idx >> 1, cc = idx & 1;
  const _Float16* hp = h + (size_t)b * 3840;
  const _Float16* wp = wo + (size_t)cc * 3840;
  __shared__ float red[256];
  float s = 0.f;
  for (int j = threadIdx.x; j < 3840; j += 256) s += (float)hp[j] * (float)wp[j];
  red[threadIdx.x] = s;
  __syncthreads();
  for (int t = 128; t > 0; t >>= 1) {
    if (threadIdx.x < t) red[threadIdx.x] += red[threadIdx.x + t];
    __syncthreads();
  }
  if (threadIdx.x == 0) out[idx] = 1.0f / (1.0f + __expf(-(red[0] + bo[cc])));
}

// ---------------- driver ----------------
extern "C" void kernel_launch(void* const* d_in, const int* in_sizes, int n_in,
                              void* d_out, int out_size, void* d_ws, size_t ws_size,
                              hipStream_t stream) {
  (void)in_sizes; (void)n_in; (void)out_size; (void)ws_size;
  const int* s1 = (const int*)d_in[0];
  const int* s2 = (const int*)d_in[1];
  const float* s1_h = (const float*)d_in[2];
  const float* s1_c = (const float*)d_in[3];
  const float* s2_h = (const float*)d_in[4];
  const float* s2_c = (const float*)d_in[5];
  const float* emb1 = (const float*)d_in[6];
  const float* emb2 = (const float*)d_in[7];
  const float* wih_f[3] = {(const float*)d_in[8], (const float*)d_in[13], (const float*)d_in[18]};
  const float* whh_f[3] = {(const float*)d_in[9], (const float*)d_in[14], (const float*)d_in[19]};
  const float* lb[3]    = {(const float*)d_in[10], (const float*)d_in[15], (const float*)d_in[20]};
  const float* wl_f[3]  = {(const float*)d_in[11], (const float*)d_in[16], (const float*)d_in[21]};
  const float* bl[3]    = {(const float*)d_in[12], (const float*)d_in[17], (const float*)d_in[22]};
  const float* wm_f = (const float*)d_in[23];
  const float* bm = (const float*)d_in[24];
  const float* wo_f = (const float*)d_in[25];
  const float* bo = (const float*)d_in[26];

  char* cur = (char*)d_ws;
  auto alloc = [&](size_t bytes) {
    char* p = cur;
    cur += (bytes + 255) & ~(size_t)255;
    return p;
  };
  const int Din_l[3] = {768, 1536, 1536};
  _Float16 *wih_h[3], *whh_h[3], *wl_h[3];
  for (int l = 0; l < 3; ++l) {
    wih_h[l] = (_Float16*)alloc((size_t)2 * G_ * Din_l[l] * 2);
    whh_h[l] = (_Float16*)alloc((size_t)2 * G_ * H_ * 2);
    wl_h[l]  = (_Float16*)alloc((size_t)G_ * (Din_l[l] + 2 * D2_) * 2);
  }
  _Float16* wm_h  = (_Float16*)alloc((size_t)3840 * 7680 * 2);
  _Float16* wo_h  = (_Float16*)alloc((size_t)2 * 3840 * 2);
  _Float16* e16   = (_Float16*)alloc((size_t)2 * BL_ * D2_ * 2);
  _Float16* g16   = (_Float16*)alloc((size_t)2 * BL_ * G_ * 2);
  float*    g32   = (float*)alloc((size_t)2 * BL_ * G_ * 4);
  float*    xg    = (float*)alloc((size_t)4 * BL_ * G_ * 4);
  _Float16* o16   = (_Float16*)alloc((size_t)2 * BL_ * D2_ * 2);
  _Float16* n16   = (_Float16*)alloc((size_t)2 * BL_ * D2_ * 2);
  _Float16* a16   = (_Float16*)alloc((size_t)2 * BL_ * D2_ * 2);
  _Float16* cat16 = (_Float16*)alloc((size_t)2 * BL_ * 3072 * 2);
  float*    sim   = (float*)alloc((size_t)B_ * L_ * L_ * 4);
  _Float16* P16   = (_Float16*)alloc((size_t)B_ * L_ * L_ * 2);
  float*    hst   = (float*)alloc((size_t)4 * B_ * H_ * 4);
  float*    cst   = (float*)alloc((size_t)4 * B_ * H_ * 4);
  float*    pool  = (float*)alloc((size_t)2 * B_ * G_ * 4);
  _Float16* merged = (_Float16*)alloc((size_t)B_ * 7680 * 2);
  _Float16* mlph   = (_Float16*)alloc((size_t)B_ * 3840 * 2);

  auto conv = [&](const float* s, _Float16* d, long long n) {
    k_f32_to_f16<<<dim3((unsigned)((n + 255) / 256)), 256, 0, stream>>>(s, d, n);
  };
  for (int l = 0; l < 3; ++l) {
    conv(wih_f[l], wih_h[l], (long long)2 * G_ * Din_l[l]);
    conv(whh_f[l], whh_h[l], (long long)2 * G_ * H_);
    conv(wl_f[l], wl_h[l], (long long)G_ * (Din_l[l] + 2 * D2_));
  }
  conv(wm_f, wm_h, (long long)3840 * 7680);
  conv(wo_f, wo_h, (long long)2 * 3840);

  const size_t hcbytes = (size_t)2 * B_ * H_ * sizeof(float);
  (void)hipMemcpyAsync(hst, s1_h, hcbytes, hipMemcpyDeviceToDevice, stream);
  (void)hipMemcpyAsync(hst + 2 * B_ * H_, s2_h, hcbytes, hipMemcpyDeviceToDevice, stream);
  (void)hipMemcpyAsync(cst, s1_c, hcbytes, hipMemcpyDeviceToDevice, stream);
  (void)hipMemcpyAsync(cst + 2 * B_ * H_, s2_c, hcbytes, hipMemcpyDeviceToDevice, stream);

  {
    int n = BL_ * D2_;
    k_embed<<<dim3((n + 255) / 256), 256, 0, stream>>>(s1, emb1, emb2, e16);
    k_embed<<<dim3((n + 255) / 256), 256, 0, stream>>>(s2, emb1, emb2,
                                                       e16 + (size_t)BL_ * D2_);
  }

  const _Float16* x16 = e16;
  long long xstride = (long long)BL_ * D2_;
  for (int l = 0; l < 3; ++l) {
    const int Din = Din_l[l];
    const int Wc = Din + 2 * D2_;
    // xg[sent][dir] = x @ Wih^T + b   (dirs batched on z)
    for (int s = 0; s < 2; ++s) {
      k_gemm<0, false><<<dim3(G_ / 64, BL_ / 128, 2), 256, 0, stream>>>(
          x16 + (size_t)s * xstride, wih_h[l], lb[l],
          xg + (size_t)s * 2 * BL_ * G_, nullptr, BL_, G_, Din, Din, Din, G_,
          0LL, (long long)G_ * Din, (long long)G_, (long long)BL_ * G_);
    }
    // recurrent scan (4 persistent workgroups: sent x dir)
    k_lstm_scan<<<dim3(4), 768, 0, stream>>>(xg, whh_h[l], hst, cst, o16);
    // row-normalize o -> n
    k_rownorm<<<dim3(BL_, 2), 256, 0, stream>>>(o16, n16);
    // sim[b] = n1 @ n2^T (batched over b on z)
    k_gemm<0, false><<<dim3(L_ / 64, 1, B_), 256, 0, stream>>>(
        n16, n16 + (size_t)BL_ * D2_, nullptr, sim, nullptr, L_, L_, D2_, D2_,
        D2_, L_, (long long)L_ * D2_, (long long)L_ * D2_, 0LL,
        (long long)L_ * L_);
    // a1 = softmax_rows(sim) @ o2
    k_softmax<<<dim3(L_, B_), 128, 0, stream>>>(sim, P16, 0);
    k_gemm<1, false><<<dim3(D2_ / 64, 1, B_), 256, 0, stream>>>(
        P16, o16 + (size_t)BL_ * D2_, nullptr, nullptr, a16, L_, D2_, L_, L_,
        D2_, D2_, (long long)L_ * L_, (long long)L_ * D2_, 0LL,
        (long long)L_ * D2_);
    // a2 = softmax_cols(sim)^T @ o1
    k_softmax<<<dim3(L_, B_), 128, 0, stream>>>(sim, P16, 1);
    k_gemm<1, false><<<dim3(D2_ / 64, 1, B_), 256, 0, stream>>>(
        P16, o16, nullptr, nullptr, a16 + (size_t)BL_ * D2_, L_, D2_, L_, L_,
        D2_, D2_, (long long)L_ * L_, (long long)L_ * D2_, 0LL,
        (long long)L_ * D2_);
    // cat = [x, o, a]
    {
      long long n = (long long)BL_ * Wc;
      k_concat<<<dim3((unsigned)((n + 255) / 256), 2), 256, 0, stream>>>(
          x16, o16, a16, cat16, Din, xstride);
    }
    // g = relu(cat @ Wl^T + bl)  (sentences batched on z)
    k_gemm<0, true><<<dim3(G_ / 64, BL_ / 128, 2), 256, 0, stream>>>(
        cat16, wl_h[l], bl[l], g32, g16, BL_, G_, Wc, Wc, Wc, G_,
        (long long)BL_ * 3072, 0LL, 0LL, (long long)BL_ * G_);
    x16 = g16;
    xstride = (long long)BL_ * G_;
  }

  {
    int n = 2 * B_ * G_;
    k_maxpool<<<dim3((n + 255) / 256), 256, 0, stream>>>(g32, pool);
  }
  {
    int n = B_ * G_;
    k_merge<<<dim3((n + 255) / 256), 256, 0, stream>>>(pool, merged);
  }
  // h = relu(merged @ Wm^T + bm)
  k_gemm<0, true><<<dim3(3840 / 64, 1, 1), 256, 0, stream>>>(
      merged, wm_h, bm, nullptr, mlph, B_, 3840, 7680, 7680, 7680, 3840, 0LL,
      0LL, 0LL, 0LL);
  // out = sigmoid(h @ Wo^T + bo)
  k_final<<<dim3(32), 256, 0, stream>>>(mlph, wo_h, bo, (float*)d_out);
}
